// Image_adapter_70300024701396
// MI455X (gfx1250) — matrix operand order repaired and checked by reading, not compiled, not run on hardware
//
#include <hip/hip_runtime.h>
#include <hip/hip_bf16.h>

#define B_   8
#define C_   128
#define S_   2304
#define HH_  8192
#define NP_  16384
#define NCH_ 16
#define KC_  (HH_ / NCH_)

typedef __attribute__((ext_vector_type(16))) __bf16        v16bf;
typedef __attribute__((ext_vector_type(8)))  float         v8f;
typedef __attribute__((ext_vector_type(8)))  unsigned int  v8u;

__device__ __forceinline__ unsigned short f2bf(float f) {
    unsigned u = __builtin_bit_cast(unsigned, f);
    u += 0x7FFFu + ((u >> 16) & 1u);   // round-to-nearest-even
    return (unsigned short)(u >> 16);
}

__device__ __forceinline__ v8f splat8(float x) {
    v8f v;
#pragma unroll
    for (int i = 0; i < 8; ++i) v[i] = x;
    return v;
}

// A fragment: 16x32 bf16, source row-major [M][K], leading dim ld (elements).
// Layout: lane&15 = row; lanes 0-15: K k0+0..15; lanes 16-31: K k0+16..31 (matches loadB).
__device__ __forceinline__ v16bf loadA(const unsigned short* p, int ld, int r0, int k0, int lane) {
    int r  = r0 + (lane & 15);
    int kb = k0 + ((lane >> 4) << 4);
    const unsigned* q0 = (const unsigned*)(p + (size_t)r * ld + kb);
    const unsigned* q1 = (const unsigned*)(p + (size_t)r * ld + kb + 8);
    v8u u;
#pragma unroll
    for (int j = 0; j < 4; ++j) u[j] = q0[j];
#pragma unroll
    for (int j = 0; j < 4; ++j) u[4 + j] = q1[j];
    return __builtin_bit_cast(v16bf, u);
}

// B fragment: 32x16 bf16, source stored [N][K] row-major (i.e. B transposed), ld in elements.
// Lanes 0-15 hold K k0+0..15, lanes 16-31 hold K k0+16..31 (pairs per VGPR).
__device__ __forceinline__ v16bf loadB(const unsigned short* p, int ld, int n0, int k0, int lane) {
    int n  = n0 + (lane & 15);
    int kb = k0 + ((lane >> 4) << 4);
    const unsigned* q = (const unsigned*)(p + (size_t)n * ld + kb);
    v8u u;
#pragma unroll
    for (int j = 0; j < 8; ++j) u[j] = q[j];
    return __builtin_bit_cast(v16bf, u);
}

__device__ __forceinline__ v8f wmma_bf(v16bf a, v16bf b, v8f c) {
    return __builtin_amdgcn_wmma_f32_16x16x32_bf16(false, a, false, b, (short)0, c, false, false);
}

// ---------- elementwise / prep kernels ----------

__global__ void k_seq_init(const float* __restrict__ feat, float* __restrict__ seq,
                           unsigned short* __restrict__ seqb) {
    int t = blockIdx.x * 256 + threadIdx.x;
    if (t >= B_ * S_ * C_) return;
    int c = t % C_;
    int s = (t / C_) % S_;
    int b = t / (C_ * S_);
    float v = feat[((size_t)b * C_ + c) * S_ + s];
    seq[t]  = v;
    seqb[t] = f2bf(v);
}

__global__ void k_prep(const float* __restrict__ qw, const float* __restrict__ kw,
                       const float* __restrict__ vw, const float* __restrict__ e1,
                       const float* __restrict__ e2,
                       unsigned short* __restrict__ wT, unsigned short* __restrict__ e1T,
                       unsigned short* __restrict__ e2T) {
    int t = blockIdx.x * 256 + threadIdx.x;
    const int WQ = 2 * 3 * C_ * C_;
    const int WE = 3 * C_ * C_;
    if (t < WQ) {
        int l3 = t / (C_ * C_);        // layer*3 + which
        int rem = t % (C_ * C_);
        int o = rem / C_, i = rem % C_;
        int layer = l3 / 3, w = l3 % 3;
        const float* src = (w == 0 ? qw : (w == 1 ? kw : vw)) + (size_t)layer * C_ * C_;
        wT[(size_t)l3 * C_ * C_ + o * C_ + i] = f2bf(src[i * C_ + o]);   // [out][in]
    } else if (t < WQ + WE) {
        int u = t - WQ;
        int e = u / (C_ * C_);
        int rem = u % (C_ * C_);
        int d = rem / C_, c = rem % C_;
        e1T[u] = f2bf(e1[(size_t)e * C_ * C_ + c * C_ + d]);             // e1T[e][d][c]
    } else if (t < WQ + 2 * WE) {
        int u = t - WQ - WE;
        int e = u / (C_ * C_);
        int rem = u % (C_ * C_);
        int f = rem / C_, d = rem % C_;
        e2T[u] = f2bf(e2[(size_t)e * C_ * C_ + d * C_ + f]);             // e2T[e][f][d]
    }
}

// ---------- hypernetwork (memory-bound, VALU) ----------

__global__ void k_h(const float* __restrict__ cond, const float* __restrict__ hw1,
                    const float* __restrict__ hb1, float* __restrict__ hbuf) {
    int n = blockIdx.x * 256 + threadIdx.x;
    if (n >= HH_) return;
    float a[B_];
    float b0 = hb1[n];
#pragma unroll
    for (int b = 0; b < B_; ++b) a[b] = b0;
    for (int k = 0; k < C_; ++k) {
        float w = hw1[(size_t)k * HH_ + n];
#pragma unroll
        for (int b = 0; b < B_; ++b) a[b] += cond[b * C_ + k] * w;
    }
#pragma unroll
    for (int b = 0; b < B_; ++b) hbuf[(size_t)b * HH_ + n] = fmaxf(a[b], 0.f);
}

__global__ void k_wkpart(const float* __restrict__ hbuf, const float* __restrict__ hw2,
                         float* __restrict__ part) {
    int t = blockIdx.x * 256 + threadIdx.x;
    if (t >= NCH_ * NP_) return;
    int n = t % NP_, ch = t / NP_;
    int k0 = ch * KC_;
    float a[B_];
#pragma unroll
    for (int b = 0; b < B_; ++b) a[b] = 0.f;
    for (int k = k0; k < k0 + KC_; ++k) {
        float w = hw2[(size_t)k * NP_ + n];                 // coalesced stream of hw2
#pragma unroll
        for (int b = 0; b < B_; ++b) a[b] += hbuf[(size_t)b * HH_ + k] * w;  // uniform -> scalar
    }
#pragma unroll
    for (int b = 0; b < B_; ++b) part[((size_t)ch * B_ + b) * NP_ + n] = a[b];
}

__global__ void k_wkred(const float* __restrict__ part, const float* __restrict__ hb2,
                        unsigned short* __restrict__ wkb) {
    int t = blockIdx.x * 256 + threadIdx.x;
    if (t >= B_ * NP_) return;
    int n = t % NP_, b = t / NP_;
    float a = hb2[n];
#pragma unroll
    for (int ch = 0; ch < NCH_; ++ch) a += part[((size_t)ch * B_ + b) * NP_ + n];
    wkb[t] = f2bf(a);                                       // wk[b][out][in] flat
}

// ---------- QKV projection (WMMA, straight-line passes, coalesced epilogues) ----------

// project seq-tile with one weight, stage bf16 in LDS, emit coalesced b128 stores
__device__ __forceinline__ void qkv_pass_rowmajor(
        const v16bf* af, const unsigned short* Wp, const float* bias,
        unsigned short* dst, unsigned short (*Tl)[136],
        int batch, int m0, int lane, int hh, int ln) {
#pragma unroll
    for (int nt = 0; nt < 8; ++nt) {
        v16bf bf[4];
#pragma unroll
        for (int kt = 0; kt < 4; ++kt) bf[kt] = loadB(Wp, C_, nt * 16, kt * 32, lane);
        v8f acc = splat8(bias[nt * 16 + ln]);
#pragma unroll
        for (int kt = 0; kt < 4; ++kt) acc = wmma_bf(af[kt], bf[kt], acc);
#pragma unroll
        for (int r = 0; r < 8; ++r)
            Tl[r + 8 * hh][nt * 16 + ln] = f2bf(acc[r]);
    }
    int row = lane >> 1, ch = (lane & 1) * 64;
    const uint4* src = (const uint4*)&Tl[row][ch];
    unsigned short* d = dst + ((size_t)batch * S_ + m0 + row) * C_ + ch;
#pragma unroll
    for (int j = 0; j < 8; ++j) ((uint4*)d)[j] = src[j];
}

__global__ __launch_bounds__(256) void k_qkv(
        const unsigned short* __restrict__ seqb, const unsigned short* __restrict__ wT,
        const float* __restrict__ qb, const float* __restrict__ kb, const float* __restrict__ vb,
        unsigned short* __restrict__ Q, unsigned short* __restrict__ K,
        unsigned short* __restrict__ Vt) {
    __shared__ __align__(16) unsigned short T[8][16][136];
    int lane = threadIdx.x & 31, wave = threadIdx.x >> 5;
    int mt = blockIdx.x * 8 + wave;
    int batch = mt / (S_ / 16);
    int m0 = (mt % (S_ / 16)) * 16;
    int hh = lane >> 4, ln = lane & 15;
    unsigned short (*Tl)[136] = T[wave];
    const unsigned short* A = seqb + (size_t)batch * S_ * C_;
    v16bf af[4];
#pragma unroll
    for (int kt = 0; kt < 4; ++kt) af[kt] = loadA(A, C_, m0, kt * 32, lane);

    // Q and K: row-major outputs via LDS repack
    qkv_pass_rowmajor(af, wT,              qb, Q, Tl, batch, m0, lane, hh, ln);
    qkv_pass_rowmajor(af, wT + C_ * C_,    kb, K, Tl, batch, m0, lane, hh, ln);

    // V: transposed output, 8 consecutive s per lane -> packed b128 stores
#pragma unroll
    for (int nt = 0; nt < 8; ++nt) {
        v16bf bf[4];
#pragma unroll
        for (int kt = 0; kt < 4; ++kt) bf[kt] = loadB(wT + 2 * C_ * C_, C_, nt * 16, kt * 32, lane);
        v8f acc = splat8(vb[nt * 16 + ln]);
#pragma unroll
        for (int kt = 0; kt < 4; ++kt) acc = wmma_bf(af[kt], bf[kt], acc);
        uint4 vv;
        vv.x = (unsigned)f2bf(acc[0]) | ((unsigned)f2bf(acc[1]) << 16);
        vv.y = (unsigned)f2bf(acc[2]) | ((unsigned)f2bf(acc[3]) << 16);
        vv.z = (unsigned)f2bf(acc[4]) | ((unsigned)f2bf(acc[5]) << 16);
        vv.w = (unsigned)f2bf(acc[6]) | ((unsigned)f2bf(acc[7]) << 16);
        int c = nt * 16 + ln;
        *(uint4*)(Vt + ((size_t)batch * C_ + c) * S_ + m0 + 8 * hh) = vv;
    }
}

// ---------- flash attention (WMMA, online softmax, LDS-staged K/V) ----------

__global__ __launch_bounds__(256) void k_attn(
        const unsigned short* __restrict__ Q, const unsigned short* __restrict__ Km,
        const unsigned short* __restrict__ Vt,
        float* __restrict__ seq, unsigned short* __restrict__ seqb) {
    __shared__ __align__(16) unsigned short Kt[128][136];   // K tile  [t_local][c]
    __shared__ __align__(16) unsigned short Vts[128][136];  // Vt tile [c][t_local]
    __shared__ __align__(16) unsigned short P[8][16][136];  // per-wave probs
    int lane = threadIdx.x & 31, wave = threadIdx.x >> 5;
    int mt = blockIdx.x * 8 + wave;                         // 8 waves share one batch
    int batch = mt / (S_ / 16);
    int m0 = (mt % (S_ / 16)) * 16;
    int hh = lane >> 4, ln = lane & 15;
    const float scale = 0.0883883476483184f;                // 1/sqrt(128)
    const unsigned short* Qb = Q  + (size_t)batch * S_ * C_;
    const unsigned short* Kb = Km + (size_t)batch * S_ * C_;
    const unsigned short* Vb = Vt + (size_t)batch * C_ * S_;
    unsigned short (*Pl)[136] = P[wave];

    int srow = threadIdx.x >> 1;
    int scol = (threadIdx.x & 1) * 64;

    v16bf qf[4];
#pragma unroll
    for (int kt = 0; kt < 4; ++kt) qf[kt] = loadA(Qb, C_, m0, kt * 32, lane);

    v8f acc[8];
    float mrow[8], lrow[8];
#pragma unroll
    for (int nt = 0; nt < 8; ++nt) acc[nt] = splat8(0.f);
#pragma unroll
    for (int r = 0; r < 8; ++r) { mrow[r] = -1e30f; lrow[r] = 0.f; }

    for (int kv0 = 0; kv0 < S_; kv0 += 128) {
        // ---- stage K and Vt tiles into LDS (whole block) ----
        {
            const uint4* gk = (const uint4*)(Kb + (size_t)(kv0 + srow) * C_ + scol);
            const uint4* gv = (const uint4*)(Vb + (size_t)srow * S_ + kv0 + scol);
            uint4* lk = (uint4*)&Kt[srow][scol];
            uint4* lv = (uint4*)&Vts[srow][scol];
#pragma unroll
            for (int j = 0; j < 8; ++j) lk[j] = gk[j];
#pragma unroll
            for (int j = 0; j < 8; ++j) lv[j] = gv[j];
            if (kv0 + 128 < S_) {  // prefetch next tile (global_prefetch_b8)
                __builtin_prefetch(Kb + (size_t)(kv0 + 128 + srow) * C_ + scol, 0, 0);
                __builtin_prefetch(Vb + (size_t)srow * S_ + kv0 + 128 + scol, 0, 0);
            }
        }
        __syncthreads();

        // ---- scores: Q @ K^T from LDS ----
        v8f st[8];
#pragma unroll
        for (int nt = 0; nt < 8; ++nt) {
            v16bf bf[4];
#pragma unroll
            for (int kt = 0; kt < 4; ++kt) bf[kt] = loadB(&Kt[0][0], 136, nt * 16, kt * 32, lane);
            v8f s = splat8(0.f);
#pragma unroll
            for (int kt = 0; kt < 4; ++kt) s = wmma_bf(qf[kt], bf[kt], s);
            st[nt] = s;
        }
        // ---- online softmax ----
        float nm[8];
#pragma unroll
        for (int r = 0; r < 8; ++r) nm[r] = mrow[r];
#pragma unroll
        for (int nt = 0; nt < 8; ++nt)
#pragma unroll
            for (int r = 0; r < 8; ++r) {
                float v = st[nt][r] * scale;
                st[nt][r] = v;
                nm[r] = fmaxf(nm[r], v);
            }
#pragma unroll
        for (int r = 0; r < 8; ++r)
#pragma unroll
            for (int msk = 1; msk < 16; msk <<= 1)
                nm[r] = fmaxf(nm[r], __shfl_xor(nm[r], msk, 32));

        float alpha[8], ls[8];
#pragma unroll
        for (int r = 0; r < 8; ++r) {
            alpha[r] = __expf(mrow[r] - nm[r]);
            mrow[r]  = nm[r];
            ls[r]    = 0.f;
        }
#pragma unroll
        for (int nt = 0; nt < 8; ++nt)
#pragma unroll
            for (int r = 0; r < 8; ++r) {
                float p = __expf(st[nt][r] - mrow[r]);
                ls[r] += p;
                Pl[r + 8 * hh][nt * 16 + ln] = f2bf(p);
            }
#pragma unroll
        for (int r = 0; r < 8; ++r) {
#pragma unroll
            for (int msk = 1; msk < 16; msk <<= 1) ls[r] += __shfl_xor(ls[r], msk, 32);
            lrow[r] = lrow[r] * alpha[r] + ls[r];
#pragma unroll
            for (int nt = 0; nt < 8; ++nt) acc[nt][r] *= alpha[r];
        }
        // ---- P @ V : A frags hoisted, B from LDS Vt tile ----
        v16bf pf[4];
#pragma unroll
        for (int kt = 0; kt < 4; ++kt) pf[kt] = loadA(&Pl[0][0], 136, 0, kt * 32, lane);
#pragma unroll
        for (int nt = 0; nt < 8; ++nt) {
            v16bf bf[4];
#pragma unroll
            for (int kt = 0; kt < 4; ++kt) bf[kt] = loadB(&Vts[0][0], 136, nt * 16, kt * 32, lane);
#pragma unroll
            for (int kt = 0; kt < 4; ++kt) acc[nt] = wmma_bf(pf[kt], bf[kt], acc[nt]);
        }
        __syncthreads();   // protect Kt/Vts before next iteration overwrites
    }
#pragma unroll
    for (int nt = 0; nt < 8; ++nt)
#pragma unroll
        for (int r = 0; r < 8; ++r) {
            int row = m0 + r + 8 * hh, c = nt * 16 + ln;
            size_t idx = ((size_t)batch * S_ + row) * C_ + c;
            float nv = seq[idx] + acc[nt][r] / lrow[r];
            seq[idx]  = nv;
            seqb[idx] = f2bf(nv);
        }
}

// ---------- MoE gate ----------

__global__ void k_gate(const float* __restrict__ seq, const float* __restrict__ gw,
                       const float* __restrict__ gb, float* __restrict__ gate) {
    int t = blockIdx.x * 256 + threadIdx.x;
    if (t >= B_ * S_) return;
    float a0 = gb[0], a1 = gb[1], a2 = gb[2];
    const float* sp = seq + (size_t)t * C_;
    for (int c = 0; c < C_; ++c) {
        float v = sp[c];
        a0 += v * gw[c * 3 + 0];
        a1 += v * gw[c * 3 + 1];
        a2 += v * gw[c * 3 + 2];
    }
    float m = fmaxf(a0, fmaxf(a1, a2));
    float e0 = __expf(a0 - m), e1 = __expf(a1 - m), e2 = __expf(a2 - m);
    float inv = 1.f / (e0 + e1 + e2);
    gate[t * 3 + 0] = e0 * inv;
    gate[t * 3 + 1] = e1 * inv;
    gate[t * 3 + 2] = e2 * inv;
}

// ---------- MoE experts (WMMA) + gated residual; writes combined ----------

__global__ __launch_bounds__(256) void k_moe(
        const unsigned short* __restrict__ seqb, const float* __restrict__ seq,
        const unsigned short* __restrict__ e1T, const unsigned short* __restrict__ e2T,
        const float* __restrict__ eb1, const float* __restrict__ eb2,
        const float* __restrict__ gate,
        float* __restrict__ out, unsigned short* __restrict__ combb) {
    __shared__ __align__(16) unsigned short P[8][16][136];
    int lane = threadIdx.x & 31, wave = threadIdx.x >> 5;
    int mt = blockIdx.x * 8 + wave;
    int batch = mt / (S_ / 16);
    int m0 = (mt % (S_ / 16)) * 16;
    int hh = lane >> 4, ln = lane & 15;
    const unsigned short* A = seqb + (size_t)batch * S_ * C_;
    unsigned short (*Pl)[136] = P[wave];

    v16bf af[4];
#pragma unroll
    for (int kt = 0; kt < 4; ++kt) af[kt] = loadA(A, C_, m0, kt * 32, lane);

    v8f moe[8];
#pragma unroll
    for (int nt = 0; nt < 8; ++nt) moe[nt] = splat8(0.f);

    float g[3][8];
#pragma unroll
    for (int r = 0; r < 8; ++r) {
        int row = m0 + r + 8 * hh;
        const float* gp = gate + ((size_t)batch * S_ + row) * 3;
        g[0][r] = gp[0]; g[1][r] = gp[1]; g[2][r] = gp[2];
    }
    for (int e = 0; e < 3; ++e) {
        const unsigned short* W1 = e1T + (size_t)e * C_ * C_;
        const unsigned short* W2 = e2T + (size_t)e * C_ * C_;
#pragma unroll
        for (int nt = 0; nt < 8; ++nt) {
            v16bf bf[4];
#pragma unroll
            for (int kt = 0; kt < 4; ++kt) bf[kt] = loadB(W1, C_, nt * 16, kt * 32, lane);
            v8f ha = splat8(eb1[e * C_ + nt * 16 + ln]);
#pragma unroll
            for (int kt = 0; kt < 4; ++kt) ha = wmma_bf(af[kt], bf[kt], ha);
#pragma unroll
            for (int r = 0; r < 8; ++r)
                Pl[r + 8 * hh][nt * 16 + ln] = f2bf(fmaxf(ha[r], 0.f));   // ReLU -> LDS
        }
        v16bf pf[4];                                   // hoisted A frags of h1
#pragma unroll
        for (int kt = 0; kt < 4; ++kt) pf[kt] = loadA(&Pl[0][0], 136, 0, kt * 32, lane);
#pragma unroll
        for (int nt = 0; nt < 8; ++nt) {
            v16bf bf[4];
#pragma unroll
            for (int kt = 0; kt < 4; ++kt) bf[kt] = loadB(W2, C_, nt * 16, kt * 32, lane);
            v8f ea = splat8(eb2[e * C_ + nt * 16 + ln]);
#pragma unroll
            for (int kt = 0; kt < 4; ++kt) ea = wmma_bf(pf[kt], bf[kt], ea);
#pragma unroll
            for (int r = 0; r < 8; ++r) moe[nt][r] += g[e][r] * ea[r];
        }
    }
    // epilogue: out[b][c][s] contiguous over r -> float4 pairs; combb repacked via LDS
#pragma unroll
    for (int nt = 0; nt < 8; ++nt) {
        int c = nt * 16 + ln;
        float cv[8];
#pragma unroll
        for (int r = 0; r < 8; ++r) {
            int row = m0 + r + 8 * hh;
            size_t si = ((size_t)batch * S_ + row) * C_ + c;
            float m  = moe[nt][r];
            float sg = 1.f / (1.f + __expf(-m));
            cv[r] = sg * m + (1.f - sg) * seq[si];
            Pl[r + 8 * hh][c] = f2bf(cv[r]);
        }
        float4 f0, f1;
        f0.x = cv[0]; f0.y = cv[1]; f0.z = cv[2]; f0.w = cv[3];
        f1.x = cv[4]; f1.y = cv[5]; f1.z = cv[6]; f1.w = cv[7];
        float* op = out + ((size_t)batch * C_ + c) * S_ + m0 + 8 * hh;
        ((float4*)op)[0] = f0;
        ((float4*)op)[1] = f1;
    }
    {
        int row = lane >> 1, ch = (lane & 1) * 64;
        const uint4* src = (const uint4*)&Pl[row][ch];
        unsigned short* dst = combb + ((size_t)batch * S_ + m0 + row) * C_ + ch;
#pragma unroll
        for (int j = 0; j < 8; ++j) ((uint4*)dst)[j] = src[j];
    }
}

// ---------- per-sample 1x1 conv (WMMA, transposed mapping), accumulate into d_out ----------

__global__ __launch_bounds__(256) void k_conv(
        const unsigned short* __restrict__ wkb, const unsigned short* __restrict__ combb,
        float* __restrict__ out) {
    int lane = threadIdx.x & 31, wave = threadIdx.x >> 5;
    int mt = blockIdx.x * 8 + wave;
    int batch = mt / (S_ / 16);
    int m0 = (mt % (S_ / 16)) * 16;                 // s-tile
    int hh = lane >> 4, ln = lane & 15;
    const unsigned short* A  = combb + (size_t)batch * S_ * C_;   // combined[b][s][i] -> A
    const unsigned short* Bw = wkb   + (size_t)batch * C_ * C_;   // wk[b][o][i] -> B rows
    v16bf af[4];
#pragma unroll
    for (int kt = 0; kt < 4; ++kt) af[kt] = loadA(A, C_, m0, kt * 32, lane);
#pragma unroll
    for (int nt = 0; nt < 8; ++nt) {                // o-tile
        v16bf bf[4];
#pragma unroll
        for (int kt = 0; kt < 4; ++kt) bf[kt] = loadB(Bw, C_, nt * 16, kt * 32, lane);
        v8f a = splat8(0.f);
#pragma unroll
        for (int kt = 0; kt < 4; ++kt) a = wmma_bf(af[kt], bf[kt], a);
        // D rows = s (contiguous over r), col = o : float4 RMW pairs
        int o = nt * 16 + ln;
        float* op = out + ((size_t)batch * C_ + o) * S_ + m0 + 8 * hh;
        float4 x0 = ((float4*)op)[0], x1 = ((float4*)op)[1];
        x0.x += a[0]; x0.y += a[1]; x0.z += a[2]; x0.w += a[3];
        x1.x += a[4]; x1.y += a[5]; x1.z += a[6]; x1.w += a[7];
        ((float4*)op)[0] = x0;
        ((float4*)op)[1] = x1;
    }
}

// ---------- host launch ----------

extern "C" void kernel_launch(void* const* d_in, const int* in_sizes, int n_in,
                              void* d_out, int out_size, void* d_ws, size_t ws_size,
                              hipStream_t stream) {
    (void)in_sizes; (void)n_in; (void)out_size; (void)ws_size;
    const float* feature = (const float*)d_in[0];
    const float* cond    = (const float*)d_in[1];
    const float* qw  = (const float*)d_in[2];
    const float* kw  = (const float*)d_in[3];
    const float* vw  = (const float*)d_in[4];
    const float* qb  = (const float*)d_in[5];
    const float* kb  = (const float*)d_in[6];
    const float* vb  = (const float*)d_in[7];
    const float* ew1 = (const float*)d_in[8];
    const float* eb1 = (const float*)d_in[9];
    const float* ew2 = (const float*)d_in[10];
    const float* eb2 = (const float*)d_in[11];
    const float* gw  = (const float*)d_in[12];
    const float* gb  = (const float*)d_in[13];
    const float* hw1 = (const float*)d_in[14];
    const float* hb1 = (const float*)d_in[15];
    const float* hw2 = (const float*)d_in[16];
    const float* hb2 = (const float*)d_in[17];
    float* out = (float*)d_out;

    char* ws = (char*)d_ws;
    size_t off = 0;
    auto alloc = [&](size_t bytes) -> char* {
        char* p = ws + off;
        off = (off + bytes + 255) & ~(size_t)255;
        return p;
    };
    float*          seq   = (float*)alloc((size_t)B_ * S_ * C_ * 4);
    unsigned short* seqb  = (unsigned short*)alloc((size_t)B_ * S_ * C_ * 2);
    unsigned short* Qb    = (unsigned short*)alloc((size_t)B_ * S_ * C_ * 2);
    unsigned short* Kbuf  = (unsigned short*)alloc((size_t)B_ * S_ * C_ * 2);
    unsigned short* Vtb   = (unsigned short*)alloc((size_t)B_ * S_ * C_ * 2);
    unsigned short* combb = (unsigned short*)alloc((size_t)B_ * S_ * C_ * 2);
    unsigned short* wT    = (unsigned short*)alloc((size_t)2 * 3 * C_ * C_ * 2);
    unsigned short* e1T   = (unsigned short*)alloc((size_t)3 * C_ * C_ * 2);
    unsigned short* e2T   = (unsigned short*)alloc((size_t)3 * C_ * C_ * 2);
    float*          gateb = (float*)alloc((size_t)B_ * S_ * 3 * 4);
    float*          hbuf  = (float*)alloc((size_t)B_ * HH_ * 4);
    float*          part  = (float*)alloc((size_t)NCH_ * B_ * NP_ * 4);
    unsigned short* wkb   = (unsigned short*)alloc((size_t)B_ * NP_ * 2);

    dim3 blk(256);
    k_seq_init<<<(B_ * S_ * C_ + 255) / 256, blk, 0, stream>>>(feature, seq, seqb);
    k_prep<<<(4 * 3 * C_ * C_ + 255) / 256, blk, 0, stream>>>(qw, kw, vw, ew1, ew2, wT, e1T, e2T);

    // hypernetwork (independent of attention path)
    k_h<<<(HH_ + 255) / 256, blk, 0, stream>>>(cond, hw1, hb1, hbuf);
    k_wkpart<<<(NCH_ * NP_ + 255) / 256, blk, 0, stream>>>(hbuf, hw2, part);
    k_wkred<<<(B_ * NP_ + 255) / 256, blk, 0, stream>>>(part, hb2, wkb);

    for (int l = 0; l < 2; ++l) {
        k_qkv<<<144, blk, 0, stream>>>(seqb, wT + (size_t)l * 3 * C_ * C_,
                                       qb + l * C_, kb + l * C_, vb + l * C_,
                                       Qb, Kbuf, Vtb);
        k_attn<<<144, blk, 0, stream>>>(Qb, Kbuf, Vtb, seq, seqb);
    }
    k_gate<<<(B_ * S_ + 255) / 256, blk, 0, stream>>>(seq, gw, gb, gateb);
    k_moe<<<144, blk, 0, stream>>>(seqb, seq, e1T, e2T, eb1, eb2, gateb, out, combb);
    k_conv<<<144, blk, 0, stream>>>(wkb, combb, out);
}
